// MultiheadAttention_29858612641871
// MI455X (gfx1250) — matrix operand order repaired and checked by reading, not compiled, not run on hardware
//
#include <hip/hip_runtime.h>
#include <hip/hip_bf16.h>

#define B_ 4
#define L_ 2048
#define E_ 1024
#define H_ 16
#define D_ 64

typedef __bf16 v16bf __attribute__((ext_vector_type(16)));
typedef float  v8f   __attribute__((ext_vector_type(8)));
typedef unsigned int u32x4 __attribute__((ext_vector_type(4)));
typedef int          i32x4 __attribute__((ext_vector_type(4)));
typedef int          i32x8 __attribute__((ext_vector_type(8)));

#if defined(__has_builtin)
#if __has_builtin(__builtin_amdgcn_tensor_load_to_lds)
#define HAVE_TDM 1
#endif
#endif

union BF16x16 {
  v16bf v;
  uint4 q[2];
  unsigned short u[16];
};

union BF16x8 {
  __bf16 h[8];
  uint4  q;
};

__device__ __forceinline__ v8f wmma_bf16(const BF16x16& a, const BF16x16& b, v8f c) {
  return __builtin_amdgcn_wmma_f32_16x16x32_bf16(false, a.v, false, b.v,
                                                 (short)0, c, false, false);
}

__device__ __forceinline__ void store8bf(unsigned short* p, v8f c) {
  BF16x8 r;
#pragma unroll
  for (int i = 0; i < 8; ++i) r.h[i] = (__bf16)c[i];
  *(uint4*)p = r.q;
}

#ifdef HAVE_TDM
// Issue a 2-D TDM tile load (bf16 elements) global -> LDS.
// D# packing per CDNA5 ISA 8.3/8.4: group0 = {count/type, lds_addr, global_addr},
// group1 = {data_size, tensor dims, tile dims, dim0 stride}. Groups 2/3 unused (2-D).
__device__ __forceinline__ void tdm_load_2d(unsigned ldsAddr, const void* gaddr,
                                            unsigned tensorDim0, unsigned tensorDim1,
                                            unsigned tileDim0, unsigned tileDim1,
                                            unsigned stride0 /*elems*/) {
  unsigned long long ga = (unsigned long long)gaddr;
  u32x4 g0;
  g0[0] = 1u;                                                  // count=1, user D#
  g0[1] = ldsAddr;                                             // LDS byte address
  g0[2] = (unsigned)(ga & 0xFFFFFFFFu);                        // global_addr[31:0]
  g0[3] = (unsigned)((ga >> 32) & 0x01FFFFFFu) | (2u << 30);   // addr[56:32] | type=2
  i32x8 g1;
  g1[0] = (int)(1u << 16);                                     // data_size=2B, mask=0
  g1[1] = (int)((tensorDim0 & 0xFFFFu) << 16);                 // tensor_dim0[15:0]
  g1[2] = (int)(((tensorDim0 >> 16) & 0xFFFFu) |
                ((tensorDim1 & 0xFFFFu) << 16));               // dim0[31:16] | dim1[15:0]
  g1[3] = (int)(((tensorDim1 >> 16) & 0xFFFFu) |
                ((tileDim0 & 0xFFFFu) << 16));                 // dim1[31:16] | tile_dim0
  g1[4] = (int)(tileDim1 & 0xFFFFu);                           // tile_dim1, tile_dim2=0
  g1[5] = (int)stride0;                                        // tensor_dim0_stride[31:0]
  g1[6] = 0;
  g1[7] = 0;
  i32x4 z4 = {0, 0, 0, 0};
#if __has_include(<hip/amd_detail/amd_gfx1250_TDM.h>)
  i32x8 z8 = {0, 0, 0, 0, 0, 0, 0, 0};
  __builtin_amdgcn_tensor_load_to_lds(g0, g1, z4, z4, z8, 0);  // 6-arg toolchain
#else
  __builtin_amdgcn_tensor_load_to_lds(g0, g1, z4, z4, 0);      // 5-arg toolchain
#endif
}
#endif

// ---------------------------------------------------------------------------
// Kernel 1a: x [B*L, E] fp32 -> xb bf16.
// ---------------------------------------------------------------------------
__global__ __launch_bounds__(256) void xconv(const float* __restrict__ x,
                                             unsigned short* __restrict__ xb) {
  const long i = ((long)blockIdx.x * 256 + threadIdx.x) * 8;
  if (i >= (long)B_ * L_ * E_) return;
  float4 f0 = *(const float4*)(x + i);
  float4 f1 = *(const float4*)(x + i + 4);
  BF16x8 r;
  r.h[0] = (__bf16)f0.x; r.h[1] = (__bf16)f0.y; r.h[2] = (__bf16)f0.z; r.h[3] = (__bf16)f0.w;
  r.h[4] = (__bf16)f1.x; r.h[5] = (__bf16)f1.y; r.h[6] = (__bf16)f1.z; r.h[7] = (__bf16)f1.w;
  *(uint4*)(xb + i) = r.q;
}

// ---------------------------------------------------------------------------
// Kernel 1b: weights [H,E,D] fp32 -> wT [3,H,D,E] bf16 (E contiguous).
// ---------------------------------------------------------------------------
__global__ __launch_bounds__(256) void wconv(const float* __restrict__ wq,
                                             const float* __restrict__ wk,
                                             const float* __restrict__ wv,
                                             unsigned short* __restrict__ wT) {
  const long t = (long)blockIdx.x * 256 + threadIdx.x;
  const long total = 3L * H_ * D_ * E_ / 8;
  if (t >= total) return;
  const long idx = t * 8;
  const int e0 = (int)(idx % E_);
  const int d  = (int)((idx / E_) % D_);
  const int h  = (int)((idx / ((long)E_ * D_)) % H_);
  const int w  = (int)(idx / ((long)E_ * D_ * H_));
  const float* src = (w == 0) ? wq : (w == 1) ? wk : wv;
  BF16x8 r;
#pragma unroll
  for (int i = 0; i < 8; ++i)
    r.h[i] = (__bf16)src[((long)h * E_ + e0 + i) * D_ + d];
  *(uint4*)(wT + idx) = r.q;
}

// ---------------------------------------------------------------------------
// Kernel 2: QKV projection, 16x64 tile/wave, double-buffered operand loads.
// grid = (B*L/64, H, 3); block = 128.
// ---------------------------------------------------------------------------
__device__ __forceinline__ void ld_shared16(const unsigned short* p, BF16x16& t) {
  t.q[0] = *(const uint4*)(p);
  t.q[1] = *(const uint4*)(p + 8);
}
__device__ __forceinline__ void ld_atile(const unsigned short* p, BF16x16& t) {
  t.q[0] = *(const uint4*)(p);
  t.q[1] = *(const uint4*)(p + 8);
}

__global__ __launch_bounds__(128) void qkv_proj(
    const unsigned short* __restrict__ xb, const unsigned short* __restrict__ wT,
    unsigned short* __restrict__ qb, unsigned short* __restrict__ kb,
    unsigned short* __restrict__ vT) {
  const int lane = threadIdx.x & 31;
  const int wave = threadIdx.x >> 5;
  const int n    = lane & 15;
  const int hi   = lane >> 4;
  const int m0   = hi ? 8 : 0;
  const int rowBase = blockIdx.x * 64 + wave * 16;
  const int h    = blockIdx.y;
  const int z    = blockIdx.z;

  const unsigned short* wh   = wT + ((long)z * H_ + h) * D_ * E_;  // [D][E]
  const unsigned short* xrow = xb + (long)(rowBase + n) * E_;
  const unsigned short* wrow = wh + (long)n * E_;

  v8f c0 = {}, c1 = {}, c2 = {}, c3 = {};
  const int b_ = rowBase / L_;
  const int lb = rowBase % L_;

  if (z < 2) {
    // (X W)^T: A rows = d (wT), B cols = x rows. Software-pipelined.
    BF16x16 bx, a0, a1, a2, a3, nbx, na0, na1, na2, na3;
    ld_shared16(xrow + hi * 16, bx);
    ld_atile(wrow + hi * 16, a0);
    ld_atile(wrow + 16L * E_ + hi * 16, a1);
    ld_atile(wrow + 32L * E_ + hi * 16, a2);
    ld_atile(wrow + 48L * E_ + hi * 16, a3);
    for (int k0 = 0; k0 < E_ - 32; k0 += 32) {
      const int kn = k0 + 32;
      ld_shared16(xrow + kn + hi * 16, nbx);
      ld_atile(wrow + kn + hi * 16, na0);
      ld_atile(wrow + 16L * E_ + kn + hi * 16, na1);
      ld_atile(wrow + 32L * E_ + kn + hi * 16, na2);
      ld_atile(wrow + 48L * E_ + kn + hi * 16, na3);
      c0 = wmma_bf16(a0, bx, c0);
      c1 = wmma_bf16(a1, bx, c1);
      c2 = wmma_bf16(a2, bx, c2);
      c3 = wmma_bf16(a3, bx, c3);
      bx = nbx; a0 = na0; a1 = na1; a2 = na2; a3 = na3;
    }
    c0 = wmma_bf16(a0, bx, c0);
    c1 = wmma_bf16(a1, bx, c1);
    c2 = wmma_bf16(a2, bx, c2);
    c3 = wmma_bf16(a3, bx, c3);

    unsigned short* dst = (z == 0) ? qb : kb;
    unsigned short* drow = dst + (((long)b_ * H_ + h) * L_ + lb + n) * D_ + m0;
    store8bf(drow,      c0);
    store8bf(drow + 16, c1);
    store8bf(drow + 32, c2);
    store8bf(drow + 48, c3);
  } else {
    // X W: A rows = x rows, B cols = d (wT). Software-pipelined.
    BF16x16 ax, b0, b1, b2, b3, nax, nb0, nb1, nb2, nb3;
    ld_atile(xrow + hi * 16, ax);
    ld_shared16(wrow + hi * 16, b0);
    ld_shared16(wrow + 16L * E_ + hi * 16, b1);
    ld_shared16(wrow + 32L * E_ + hi * 16, b2);
    ld_shared16(wrow + 48L * E_ + hi * 16, b3);
    for (int k0 = 0; k0 < E_ - 32; k0 += 32) {
      const int kn = k0 + 32;
      ld_atile(xrow + kn + hi * 16, nax);
      ld_shared16(wrow + kn + hi * 16, nb0);
      ld_shared16(wrow + 16L * E_ + kn + hi * 16, nb1);
      ld_shared16(wrow + 32L * E_ + kn + hi * 16, nb2);
      ld_shared16(wrow + 48L * E_ + kn + hi * 16, nb3);
      c0 = wmma_bf16(ax, b0, c0);
      c1 = wmma_bf16(ax, b1, c1);
      c2 = wmma_bf16(ax, b2, c2);
      c3 = wmma_bf16(ax, b3, c3);
      ax = nax; b0 = nb0; b1 = nb1; b2 = nb2; b3 = nb3;
    }
    c0 = wmma_bf16(ax, b0, c0);
    c1 = wmma_bf16(ax, b1, c1);
    c2 = wmma_bf16(ax, b2, c2);
    c3 = wmma_bf16(ax, b3, c3);

    unsigned short* vrow = vT + (((long)b_ * H_ + h) * D_ + n) * L_ + lb + m0;
    store8bf(vrow,            c0);
    store8bf(vrow + 16L * L_, c1);
    store8bf(vrow + 32L * L_, c2);
    store8bf(vrow + 48L * L_, c3);
  }
}

// ---------------------------------------------------------------------------
// Kernel 3: causal flash attention with TDM-staged K/V double buffer in LDS.
// grid = (L/128, B*H); block = 256 (8 waves; wave w owns queries qBase..+15).
// ---------------------------------------------------------------------------
__global__ __launch_bounds__(256) void flash_attn(
    const unsigned short* __restrict__ qb, const unsigned short* __restrict__ kb,
    const unsigned short* __restrict__ vT, float* __restrict__ out) {
  __shared__ __align__(16) unsigned short Kbuf[2][32 * 64];  // [key][d]
  __shared__ __align__(16) unsigned short Vbuf[2][64 * 32];  // [d][key]
  __shared__ __align__(16) unsigned short Plds[8][16 * 32];

  const int lane = threadIdx.x & 31;
  const int wave = threadIdx.x >> 5;
  const int n    = lane & 15;
  const int hi   = lane >> 4;
  const int m0   = hi ? 8 : 0;
  const int bh   = blockIdx.y;
  const int b_   = bh / H_, h = bh % H_;
  const int qBase = blockIdx.x * 128 + wave * 16;

  const unsigned short* qp = qb + (long)bh * L_ * D_;
  const unsigned short* kp = kb + (long)bh * L_ * D_;
  const unsigned short* vp = vT + (long)bh * D_ * L_;

  // Preload Q as two 16x32 A tiles (d = 0..31 and 32..63).
  BF16x16 a0, a1;
  {
    const unsigned short* qr = qp + (long)(qBase + n) * D_ + hi * 16;
    a0.q[0] = *(const uint4*)(qr);
    a0.q[1] = *(const uint4*)(qr + 8);
    a1.q[0] = *(const uint4*)(qr + 32);
    a1.q[1] = *(const uint4*)(qr + 40);
  }

  float m_i[8], l_i[8];
  v8f o0 = {}, o1 = {}, o2 = {}, o3 = {};
#pragma unroll
  for (int r = 0; r < 8; ++r) { m_i[r] = -1e30f; l_i[r] = 0.f; }

  const float scale = 0.125f;
  const int jEnd = (qBase + 15) >> 5;     // this wave's last key block
  const int jMax = blockIdx.x * 4 + 3;    // block-uniform last key block

  // Stage key block j into buffer buf (K: 32x64 tile, V: 64x32 tile).
  auto stage = [&](int j, int buf) {
    const int key0 = j << 5;
#ifdef HAVE_TDM
    if (wave == 0) {
      tdm_load_2d((unsigned)(unsigned long long)&Kbuf[buf][0],
                  kp + (long)key0 * D_, D_, L_, D_, 32, D_);
      tdm_load_2d((unsigned)(unsigned long long)&Vbuf[buf][0],
                  vp + key0, L_, D_, 32, D_, L_);
    }
#else
    const int t = threadIdx.x;            // 256 threads, 8 bf16 each
    *(uint4*)&Kbuf[buf][t * 8] = *(const uint4*)(kp + (long)key0 * D_ + t * 8);
    const int vd = t >> 2, vc = (t & 3) * 8;
    *(uint4*)&Vbuf[buf][vd * 32 + vc] = *(const uint4*)(vp + (long)vd * L_ + key0 + vc);
#endif
  };

  stage(0, 0);

  for (int j = 0; j <= jMax; ++j) {
#ifdef HAVE_TDM
    __builtin_amdgcn_s_wait_tensorcnt(0);   // no-op for waves != 0
#endif
    __syncthreads();                        // buf[j&1] ready; prev compute done
    if (j < jMax) stage(j + 1, (j + 1) & 1);

    if (j <= jEnd) {
      const int key0 = j << 5;
      const unsigned short* kbuf = &Kbuf[j & 1][0];
      const unsigned short* vbuf = &Vbuf[j & 1][0];

      // ---- S = Q K^T (two 16-key C tiles, K-dim 64 = 2 WMMAs each) ----
      v8f s0 = {}, s1 = {};
      {
        const unsigned short* kr = kbuf + n * D_ + hi * 16;
        BF16x16 t0, t1;
        t0.q[0] = *(const uint4*)(kr);      t0.q[1] = *(const uint4*)(kr + 8);
        t1.q[0] = *(const uint4*)(kr + 32); t1.q[1] = *(const uint4*)(kr + 40);
        s0 = wmma_bf16(a0, t0, s0);
        s0 = wmma_bf16(a1, t1, s0);
        const unsigned short* kr2 = kr + 16 * D_;
        t0.q[0] = *(const uint4*)(kr2);      t0.q[1] = *(const uint4*)(kr2 + 8);
        t1.q[0] = *(const uint4*)(kr2 + 32); t1.q[1] = *(const uint4*)(kr2 + 40);
        s1 = wmma_bf16(a0, t0, s1);
        s1 = wmma_bf16(a1, t1, s1);
      }

#pragma unroll
      for (int r = 0; r < 8; ++r) { s0[r] *= scale; s1[r] *= scale; }

      if (key0 + 31 > qBase) {              // diagonal block: causal mask
#pragma unroll
        for (int r = 0; r < 8; ++r) {
          const int qi = qBase + r + m0;
          if (key0 + n      > qi) s0[r] = -1e30f;
          if (key0 + n + 16 > qi) s1[r] = -1e30f;
        }
      }

      // ---- online softmax ----
      float p0[8], p1[8];
#pragma unroll
      for (int r = 0; r < 8; ++r) {
        float mx = fmaxf(s0[r], s1[r]);
#pragma unroll
        for (int sh = 1; sh < 16; sh <<= 1)
          mx = fmaxf(mx, __shfl_xor(mx, sh, 32));
        const float mnew  = fmaxf(m_i[r], mx);
        const float alpha = __expf(m_i[r] - mnew);
        m_i[r] = mnew;
        const float e0 = __expf(s0[r] - mnew);
        const float e1 = __expf(s1[r] - mnew);
        p0[r] = e0; p1[r] = e1;
        float rs = e0 + e1;
#pragma unroll
        for (int sh = 1; sh < 16; sh <<= 1)
          rs += __shfl_xor(rs, sh, 32);
        l_i[r] = l_i[r] * alpha + rs;
        o0[r] *= alpha; o1[r] *= alpha; o2[r] *= alpha; o3[r] *= alpha;
      }

      // ---- P: C layout -> A layout via per-wave LDS round trip ----
      __bf16* pw = (__bf16*)&Plds[wave][0];
#pragma unroll
      for (int r = 0; r < 8; ++r) {
        const int m = r + m0;
        pw[m * 32 + n]      = (__bf16)p0[r];
        pw[m * 32 + n + 16] = (__bf16)p1[r];
      }
      BF16x16 pa;
      {
        const uint4* pr = (const uint4*)((unsigned short*)pw + n * 32 + hi * 16);
        pa.q[0] = pr[0];
        pa.q[1] = pr[1];
      }

      // ---- O += P * V (4 column tiles of D, K-dim = 32 keys) ----
      {
        const unsigned short* vr = vbuf + n * 32 + hi * 16;
        BF16x16 vb;
        vb.q[0] = *(const uint4*)(vr);          vb.q[1] = *(const uint4*)(vr + 8);
        o0 = wmma_bf16(pa, vb, o0);
        vb.q[0] = *(const uint4*)(vr + 512);    vb.q[1] = *(const uint4*)(vr + 520);
        o1 = wmma_bf16(pa, vb, o1);
        vb.q[0] = *(const uint4*)(vr + 1024);   vb.q[1] = *(const uint4*)(vr + 1032);
        o2 = wmma_bf16(pa, vb, o2);
        vb.q[0] = *(const uint4*)(vr + 1536);   vb.q[1] = *(const uint4*)(vr + 1544);
        o3 = wmma_bf16(pa, vb, o3);
      }
    }
  }

  // ---- epilogue: out [B, L, H*D] fp32 ----
  float* orow = out + ((long)b_ * L_ + qBase) * (H_ * D_) + h * D_;
#pragma unroll
  for (int r = 0; r < 8; ++r) {
    const int m = r + m0;
    const float inv = 1.0f / l_i[r];
    float* dst = orow + (long)m * (H_ * D_);
    dst[n]      = o0[r] * inv;
    dst[16 + n] = o1[r] * inv;
    dst[32 + n] = o2[r] * inv;
    dst[48 + n] = o3[r] * inv;
  }
}

// ---------------------------------------------------------------------------
extern "C" void kernel_launch(void* const* d_in, const int* in_sizes, int n_in,
                              void* d_out, int out_size, void* d_ws, size_t ws_size,
                              hipStream_t stream) {
  const float* x  = (const float*)d_in[0];
  const float* wq = (const float*)d_in[1];
  const float* wk = (const float*)d_in[2];
  const float* wv = (const float*)d_in[3];
  float* out = (float*)d_out;

  unsigned char* ws = (unsigned char*)d_ws;
  const size_t wTelems  = 3ull * H_ * D_ * E_;
  const size_t xElems   = (size_t)B_ * L_ * E_;
  const size_t qkvElems = (size_t)B_ * H_ * L_ * D_;
  unsigned short* wT = (unsigned short*)ws;
  unsigned short* xb = wT + wTelems;
  unsigned short* qb = xb + xElems;
  unsigned short* kb = qb + qkvElems;
  unsigned short* vT = kb + qkvElems;

  xconv<<<dim3((unsigned)((xElems / 8 + 255) / 256)), 256, 0, stream>>>(x, xb);
  wconv<<<dim3((unsigned)((wTelems / 8 + 255) / 256)), 256, 0, stream>>>(wq, wk, wv, wT);
  {
    dim3 g(B_ * L_ / 64, H_, 3);
    qkv_proj<<<g, 128, 0, stream>>>(xb, wT, qb, kb, vT);
  }
  {
    dim3 g(L_ / 128, B_ * H_);
    flash_attn<<<g, 256, 0, stream>>>(qb, kb, vT, out);
  }
}
